// Seq2Seq_68607807586578
// MI455X (gfx1250) — matrix operand order repaired and checked by reading, not compiled, not run on hardware
//
#include <hip/hip_runtime.h>
#include <math.h>

// ---- problem constants (match reference) ----
#define TT 128      // encoder length
#define SS 128      // decoder length
#define BB 32       // batch
#define HH 512      // hidden == input size
#define H4 2048     // 4*H
#define VD_ 32000   // decoder vocab

#define AS1 __attribute__((address_space(1)))
#define AS3 __attribute__((address_space(3)))

typedef __attribute__((ext_vector_type(16))) __bf16 v16bf;
typedef __attribute__((ext_vector_type(8)))  __bf16 v8bf;
typedef __attribute__((ext_vector_type(8)))  float  v8f;
typedef int vint4 __attribute__((vector_size(16)));   // matches builtin param type

// float -> bf16 (round to nearest even), bit-level
__device__ __forceinline__ unsigned short f2bf(float f) {
    unsigned int u = __builtin_bit_cast(unsigned int, f);
    u += 0x7FFFu + ((u >> 16) & 1u);
    return (unsigned short)(u >> 16);
}

// A-operand tile fragment (16x32 bf16, MxK). rowbase = base + row*stride + k0.
// Lane group g: elements K = {g*8 .. g*8+7} and {16+g*8 .. 16+g*8+7}.
__device__ __forceinline__ v16bf load_a(const unsigned short* rowbase, int g) {
    v8bf lo = *reinterpret_cast<const v8bf*>(rowbase + g * 8);
    v8bf hi = *reinterpret_cast<const v8bf*>(rowbase + 16 + g * 8);
    v16bf r;
#pragma unroll
    for (int i = 0; i < 8; ++i) { r[i] = lo[i]; r[i + 8] = hi[i]; }
    return r;
}

// B-operand tile fragment (32x16 bf16, KxN). Lane n = lane&15 holds column n.
// Lane group g: K = {g*8 .. g*8+7} and {16+g*8 .. 16+g*8+7}.
__device__ __forceinline__ v16bf load_b(const unsigned short* rowbase, int g) {
    v8bf lo = *reinterpret_cast<const v8bf*>(rowbase + g * 8);
    v8bf hi = *reinterpret_cast<const v8bf*>(rowbase + 16 + g * 8);
    v16bf r;
#pragma unroll
    for (int i = 0; i < 8; ++i) { r[i] = lo[i]; r[i + 8] = hi[i]; }
    return r;
}

__device__ __forceinline__ v8f wmma_bf16(v16bf a, v16bf b, v8f c) {
    return __builtin_amdgcn_wmma_f32_16x16x32_bf16(false, a, false, b,
                                                   (short)0, c, false, false);
}

// ---------------- utility kernels ----------------

__global__ void k_f32_to_bf16(const float* __restrict__ in,
                              unsigned short* __restrict__ out, int n) {
    int i = blockIdx.x * blockDim.x + threadIdx.x;
    if (i < n) out[i] = f2bf(in[i]);
}

// gather embeddings to bf16 activations: out[(row)*H + d] = emb[idx[row]*H + d]
__global__ void k_embed(const int* __restrict__ idx, const float* __restrict__ emb,
                        unsigned short* __restrict__ out, int nrows) {
    int gid = blockIdx.x * blockDim.x + threadIdx.x;
    if (gid >= nrows * HH) return;
    int row = gid >> 9;        // / 512
    int d = gid & 511;
    int tok = idx[row];
    out[gid] = f2bf(emb[(size_t)tok * HH + d]);
}

__global__ void k_init_state(unsigned short* __restrict__ h0, float* __restrict__ c) {
    int i = blockIdx.x * blockDim.x + threadIdx.x;
    if (i < BB * HH) { h0[i] = 0; c[i] = 0.0f; }
}

// ---------------- fused LSTM step ----------------
// grid 64 = (row-tile 0..1) x (unit-tile 0..31); block 128 = 4 waves, wave = gate (i,f,g,o).
// gates = x_t @ Wih^T + h @ Whh^T (both fused in one K loop), then elementwise update.
__global__ void k_lstm_step(const unsigned short* __restrict__ x_bf,  // [L*B, H]
                            const unsigned short* __restrict__ Wih,   // [4H, H]
                            const unsigned short* __restrict__ Whh,   // [4H, H]
                            const unsigned short* __restrict__ h_in,  // [B, H]
                            unsigned short* __restrict__ h_out,       // [B, H]
                            float* __restrict__ c,                    // [B, H]
                            unsigned short* __restrict__ seq_out,     // [L, B, H]
                            unsigned short* __restrict__ seq_outT,    // [B, H, L] | null
                            int t, int Lseq) {
    __shared__ float g_lds[4][16][16];
    const int tid = threadIdx.x;
    const int lane = tid & 31;
    const int w = tid >> 5;            // gate 0..3
    const int rt = blockIdx.x >> 5;    // batch tile
    const int ct = blockIdx.x & 31;    // unit tile
    const int gl = lane >> 4;
    const int r = lane & 15;

    const unsigned short* arow_x = x_bf + ((size_t)(t * BB + rt * 16 + r)) * HH;
    const unsigned short* arow_h = h_in + (size_t)(rt * 16 + r) * HH;
    const unsigned short* brow_i = Wih + (size_t)(w * HH + ct * 16 + r) * HH;
    const unsigned short* brow_h = Whh + (size_t)(w * HH + ct * 16 + r) * HH;

    v8f acc = {};
    for (int k0 = 0; k0 < HH; k0 += 32) {
        acc = wmma_bf16(load_a(arow_x + k0, gl), load_b(brow_i + k0, gl), acc);
        acc = wmma_bf16(load_a(arow_h + k0, gl), load_b(brow_h + k0, gl), acc);
    }
#pragma unroll
    for (int j = 0; j < 8; ++j) g_lds[w][j + 8 * gl][r] = acc[j];
    __syncthreads();

#pragma unroll
    for (int e2 = 0; e2 < 2; ++e2) {
        int e = tid * 2 + e2;
        int m = e >> 4, n = e & 15;
        int b = rt * 16 + m, j = ct * 16 + n;
        float gi = g_lds[0][m][n];
        float gf = g_lds[1][m][n];
        float gg = g_lds[2][m][n];
        float go = g_lds[3][m][n];
        float si = 1.0f / (1.0f + expf(-gi));
        float sf = 1.0f / (1.0f + expf(-gf));
        float so = 1.0f / (1.0f + expf(-go));
        float cn = sf * c[b * HH + j] + si * tanhf(gg);
        float hv = so * tanhf(cn);
        c[b * HH + j] = cn;
        unsigned short hb = f2bf(hv);
        h_out[b * HH + j] = hb;
        seq_out[((size_t)t * BB + b) * HH + j] = hb;
        if (seq_outT) seq_outT[((size_t)b * HH + j) * Lseq + t] = hb;
    }
}

// ---------------- attention ----------------
// scores[b,s,t] = dec_out[s,b,:] . enc_out[t,b,:]
__global__ void k_scores(const unsigned short* __restrict__ dec_bf,  // [S,B,H]
                         const unsigned short* __restrict__ enc_bf,  // [T,B,H]
                         float* __restrict__ scores) {               // [B,S,T]
    const int lane = threadIdx.x & 31;
    const int wid = blockIdx.x * (blockDim.x >> 5) + (threadIdx.x >> 5);
    const int b = wid >> 6;
    const int st = (wid >> 3) & 7;
    const int tt = wid & 7;
    const int gl = lane >> 4, r = lane & 15;

    const unsigned short* arow = dec_bf + ((size_t)(st * 16 + r) * BB + b) * HH;
    const unsigned short* brow = enc_bf + ((size_t)(tt * 16 + r) * BB + b) * HH;
    v8f acc = {};
    for (int k0 = 0; k0 < HH; k0 += 32)
        acc = wmma_bf16(load_a(arow + k0, gl), load_b(brow + k0, gl), acc);
#pragma unroll
    for (int j = 0; j < 8; ++j) {
        int s = st * 16 + j + 8 * gl;
        int t = tt * 16 + r;
        scores[((size_t)b * SS + s) * TT + t] = acc[j];
    }
}

// row softmax over T=128; one wave per row; output bf16
__global__ void k_softmax(const float* __restrict__ scores,
                          unsigned short* __restrict__ attn) {
    const int lane = threadIdx.x & 31;
    const int row = blockIdx.x * (blockDim.x >> 5) + (threadIdx.x >> 5);
    const float* sr = scores + (size_t)row * TT;
    float v[4];
    float m = -1e30f;
#pragma unroll
    for (int i = 0; i < 4; ++i) { v[i] = sr[lane + i * 32]; m = fmaxf(m, v[i]); }
#pragma unroll
    for (int off = 16; off > 0; off >>= 1) m = fmaxf(m, __shfl_xor(m, off, 32));
    float s = 0.f;
#pragma unroll
    for (int i = 0; i < 4; ++i) { v[i] = expf(v[i] - m); s += v[i]; }
#pragma unroll
    for (int off = 16; off > 0; off >>= 1) s += __shfl_xor(s, off, 32);
    float inv = 1.0f / s;
    unsigned short* ar = attn + (size_t)row * TT;
#pragma unroll
    for (int i = 0; i < 4; ++i) ar[lane + i * 32] = f2bf(v[i] * inv);
}

// ctx[s,b,:] = attn[b,s,:] @ enc_out[:,b,:]  (B-operand from transposed encT[b,h,t])
__global__ void k_ctx(const unsigned short* __restrict__ attn,  // [B,S,T]
                      const unsigned short* __restrict__ encT,  // [B,H,T]
                      unsigned short* __restrict__ ctx) {       // [S,B,H]
    const int lane = threadIdx.x & 31;
    const int wid = blockIdx.x * (blockDim.x >> 5) + (threadIdx.x >> 5);
    const int b = wid >> 8;
    const int st = (wid >> 5) & 7;
    const int ht = wid & 31;
    const int gl = lane >> 4, r = lane & 15;
    const unsigned short* arow = attn + (size_t)b * SS * TT + (size_t)(st * 16 + r) * TT;
    const unsigned short* brow = encT + (size_t)b * HH * TT + (size_t)(ht * 16 + r) * TT;
    v8f acc = {};
#pragma unroll
    for (int k0 = 0; k0 < TT; k0 += 32)
        acc = wmma_bf16(load_a(arow + k0, gl), load_b(brow + k0, gl), acc);
#pragma unroll
    for (int j = 0; j < 8; ++j) {
        int s = st * 16 + j + 8 * gl;
        int h = ht * 16 + r;
        ctx[((size_t)s * BB + b) * HH + h] = f2bf(acc[j]);
    }
}

// hidden = tanh(ctx @ We^T + eb + dec @ Wd^T + db), two fused GEMMs
__global__ void k_hidden(const unsigned short* __restrict__ ctx,  // [S*B, H]
                         const unsigned short* __restrict__ dec,  // [S*B, H]
                         const unsigned short* __restrict__ We,   // [H, H]
                         const unsigned short* __restrict__ Wd,   // [H, H]
                         const float* __restrict__ eb, const float* __restrict__ db,
                         unsigned short* __restrict__ hidden) {   // [S*B, H]
    const int lane = threadIdx.x & 31;
    const int wid = blockIdx.x * (blockDim.x >> 5) + (threadIdx.x >> 5);
    const int mt = wid >> 5;
    const int nt = wid & 31;
    const int gl = lane >> 4, r = lane & 15;
    const unsigned short* a1 = ctx + (size_t)(mt * 16 + r) * HH;
    const unsigned short* a2 = dec + (size_t)(mt * 16 + r) * HH;
    const unsigned short* b1 = We + (size_t)(nt * 16 + r) * HH;
    const unsigned short* b2 = Wd + (size_t)(nt * 16 + r) * HH;
    v8f acc = {};
    for (int k0 = 0; k0 < HH; k0 += 32) {
        acc = wmma_bf16(load_a(a1 + k0, gl), load_b(b1 + k0, gl), acc);
        acc = wmma_bf16(load_a(a2 + k0, gl), load_b(b2 + k0, gl), acc);
    }
    int col = nt * 16 + r;
    float bias = eb[col] + db[col];
#pragma unroll
    for (int j = 0; j < 8; ++j) {
        int row = mt * 16 + j + 8 * gl;
        hidden[(size_t)row * HH + col] = f2bf(tanhf(acc[j] + bias));
    }
}

// ---------------- output projection ----------------
// Macro-tiled: each block owns 64 rows x 128 cols. The 64x512 bf16 A slab
// (contiguous 64KB of `hidden`) is staged once into LDS via async global->LDS
// copies and shared by all 8 waves; each wave computes a 16x64 footprint.
__global__ void __launch_bounds__(256)
k_outproj(const unsigned short* __restrict__ hidden,  // [4096,512]
          const unsigned short* __restrict__ Wout,    // [32000,512]
          const float* __restrict__ ob,               // [32000]
          float* __restrict__ out) {                  // [4096,32000]
    __shared__ unsigned short As[64 * HH];            // 64 KB

    const int mb = blockIdx.x / 250;   // 0..63   (64-row block)
    const int nb = blockIdx.x % 250;   // 0..249  (128-col block)
    const int tid = threadIdx.x;
    const int lane = tid & 31;
    const int w = tid >> 5;            // wave 0..7
    const int mt = w >> 1;             // local M tile 0..3
    const int nq = w & 1;              // local N quad 0..1 (4 tiles each)
    const int gl = lane >> 4, r = lane & 15;

    // ---- stage A macro-tile into LDS (16 x 16B chunks per thread) ----
    const unsigned short* gsrc = hidden + (size_t)mb * 64 * HH;
#if __has_builtin(__builtin_amdgcn_global_load_async_to_lds_b128)
#pragma unroll
    for (int j = 0; j < 16; ++j) {
        int off = (tid + j * 256) * 8;  // ushort units, 16B granules
        __builtin_amdgcn_global_load_async_to_lds_b128(
            (AS1 vint4*)(gsrc + off), (AS3 vint4*)(As + off), 0, 0);
    }
#if __has_builtin(__builtin_amdgcn_s_wait_asynccnt)
    __builtin_amdgcn_s_wait_asynccnt(0);
#else
    asm volatile("s_wait_asynccnt 0x0" ::: "memory");
#endif
#else
#pragma unroll
    for (int j = 0; j < 16; ++j) {
        int off = (tid + j * 256) * 8;
        *reinterpret_cast<v8bf*>(As + off) =
            *reinterpret_cast<const v8bf*>(gsrc + off);
    }
#endif
    __syncthreads();

    // ---- GEMM: A from LDS, B streamed from L2 ----
    const unsigned short* arow = As + (size_t)(mt * 16 + r) * HH;
    const unsigned short* brow = Wout + (size_t)(nb * 128 + nq * 64 + r) * HH;
    v8f acc0 = {}, acc1 = {}, acc2 = {}, acc3 = {};
    for (int k0 = 0; k0 < HH; k0 += 32) {
        v16bf a = load_a(arow + k0, gl);
        acc0 = wmma_bf16(a, load_b(brow + k0, gl), acc0);
        acc1 = wmma_bf16(a, load_b(brow + 16 * HH + k0, gl), acc1);
        acc2 = wmma_bf16(a, load_b(brow + 32 * HH + k0, gl), acc2);
        acc3 = wmma_bf16(a, load_b(brow + 48 * HH + k0, gl), acc3);
    }
#pragma unroll
    for (int u = 0; u < 4; ++u) {
        v8f acc = (u == 0) ? acc0 : (u == 1) ? acc1 : (u == 2) ? acc2 : acc3;
        int col = nb * 128 + nq * 64 + u * 16 + r;
        float bias = ob[col];
#pragma unroll
        for (int j = 0; j < 8; ++j) {
            int row = mb * 64 + mt * 16 + j + 8 * gl;
            out[(size_t)row * VD_ + col] = acc[j] + bias;
        }
    }
}

// in-place log-softmax over vocab; one block per row (rows stay resident in L2)
__global__ void k_logsoftmax(float* __restrict__ out) {
    __shared__ float red[256];
    const int row = blockIdx.x;
    float* p = out + (size_t)row * VD_;
    const int tid = threadIdx.x;
    float m = -1e30f;
    for (int i = tid; i < VD_; i += 256) m = fmaxf(m, p[i]);
    red[tid] = m;
    __syncthreads();
    for (int s = 128; s > 0; s >>= 1) {
        if (tid < s) red[tid] = fmaxf(red[tid], red[tid + s]);
        __syncthreads();
    }
    m = red[0];
    __syncthreads();
    float sum = 0.f;
    for (int i = tid; i < VD_; i += 256) sum += expf(p[i] - m);
    red[tid] = sum;
    __syncthreads();
    for (int s = 128; s > 0; s >>= 1) {
        if (tid < s) red[tid] += red[tid + s];
        __syncthreads();
    }
    float lse = m + logf(red[0]);
    for (int i = tid; i < VD_; i += 256) p[i] = p[i] - lse;
}

// ---------------- host orchestration ----------------
extern "C" void kernel_launch(void* const* d_in, const int* in_sizes, int n_in,
                              void* d_out, int out_size, void* d_ws, size_t ws_size,
                              hipStream_t stream) {
    (void)in_sizes; (void)n_in; (void)out_size; (void)ws_size;
    const int* enc_idx = (const int*)d_in[0];
    const int* dec_idx = (const int*)d_in[1];
    const float* enc_emb = (const float*)d_in[2];
    const float* dec_emb = (const float*)d_in[3];
    const float* f_enc_Wih = (const float*)d_in[4];
    const float* f_enc_Whh = (const float*)d_in[5];
    const float* f_dec_Wih = (const float*)d_in[6];
    const float* f_dec_Whh = (const float*)d_in[7];
    const float* f_We = (const float*)d_in[8];
    const float* ebias = (const float*)d_in[9];
    const float* f_Wd = (const float*)d_in[10];
    const float* dbias = (const float*)d_in[11];
    const float* f_Wout = (const float*)d_in[12];
    const float* ob = (const float*)d_in[13];
    float* out = (float*)d_out;

    char* ws = (char*)d_ws;
    size_t off = 0;
    auto carve = [&](size_t bytes) -> char* {
        char* p = ws + off;
        off += (bytes + 255) & ~(size_t)255;
        return p;
    };
    unsigned short* wih_e = (unsigned short*)carve((size_t)H4 * HH * 2);
    unsigned short* whh_e = (unsigned short*)carve((size_t)H4 * HH * 2);
    unsigned short* wih_d = (unsigned short*)carve((size_t)H4 * HH * 2);
    unsigned short* whh_d = (unsigned short*)carve((size_t)H4 * HH * 2);
    unsigned short* We = (unsigned short*)carve((size_t)HH * HH * 2);
    unsigned short* Wd = (unsigned short*)carve((size_t)HH * HH * 2);
    unsigned short* Wout = (unsigned short*)carve((size_t)VD_ * HH * 2);
    unsigned short* xe = (unsigned short*)carve((size_t)TT * BB * HH * 2);
    unsigned short* xd = (unsigned short*)carve((size_t)SS * BB * HH * 2);
    unsigned short* hA = (unsigned short*)carve((size_t)BB * HH * 2);
    unsigned short* hB = (unsigned short*)carve((size_t)BB * HH * 2);
    float* cbuf = (float*)carve((size_t)BB * HH * 4);
    unsigned short* enc_out = (unsigned short*)carve((size_t)TT * BB * HH * 2);
    unsigned short* encT = (unsigned short*)carve((size_t)BB * HH * TT * 2);
    unsigned short* dec_out = (unsigned short*)carve((size_t)SS * BB * HH * 2);
    float* scores = (float*)carve((size_t)BB * SS * TT * 4);
    unsigned short* attn = (unsigned short*)carve((size_t)BB * SS * TT * 2);
    unsigned short* ctx = (unsigned short*)carve((size_t)SS * BB * HH * 2);
    unsigned short* hidden = (unsigned short*)carve((size_t)SS * BB * HH * 2);

    auto cvt = [&](const float* src, unsigned short* dst, int n) {
        k_f32_to_bf16<<<(n + 255) / 256, 256, 0, stream>>>(src, dst, n);
    };
    cvt(f_enc_Wih, wih_e, H4 * HH);
    cvt(f_enc_Whh, whh_e, H4 * HH);
    cvt(f_dec_Wih, wih_d, H4 * HH);
    cvt(f_dec_Whh, whh_d, H4 * HH);
    cvt(f_We, We, HH * HH);
    cvt(f_Wd, Wd, HH * HH);
    cvt(f_Wout, Wout, VD_ * HH);

    k_embed<<<(TT * BB * HH + 255) / 256, 256, 0, stream>>>(enc_idx, enc_emb, xe, TT * BB);
    k_embed<<<(SS * BB * HH + 255) / 256, 256, 0, stream>>>(dec_idx, dec_emb, xd, SS * BB);
    k_init_state<<<(BB * HH + 255) / 256, 256, 0, stream>>>(hA, cbuf);

    // 128 encoder + 128 decoder steps; decoder inherits (h, c) from encoder
    for (int u = 0; u < TT + SS; ++u) {
        bool enc = (u < TT);
        int t = enc ? u : u - TT;
        unsigned short* hin = (u & 1) ? hB : hA;
        unsigned short* hout = (u & 1) ? hA : hB;
        k_lstm_step<<<64, 128, 0, stream>>>(
            enc ? xe : xd, enc ? wih_e : wih_d, enc ? whh_e : whh_d,
            hin, hout, cbuf,
            enc ? enc_out : dec_out,
            enc ? encT : (unsigned short*)nullptr, t, TT);
    }

    k_scores<<<(BB * 8 * 8) / 8, 256, 0, stream>>>(dec_out, enc_out, scores);
    k_softmax<<<(BB * SS) / 8, 256, 0, stream>>>(scores, attn);
    k_ctx<<<(BB * 8 * 32) / 8, 256, 0, stream>>>(attn, encT, ctx);
    k_hidden<<<(256 * 32) / 8, 256, 0, stream>>>(ctx, dec_out, We, Wd, ebias, dbias, hidden);
    k_outproj<<<64 * 250, 256, 0, stream>>>(hidden, Wout, ob, out);
    k_logsoftmax<<<SS * BB, 256, 0, stream>>>(out);
}
